// OutBlock_1881195675665
// MI455X (gfx1250) — matrix operand order repaired and checked by reading, not compiled, not run on hardware
//
#include <hip/hip_runtime.h>

typedef __attribute__((ext_vector_type(16))) _Float16 v16h;
typedef __attribute__((ext_vector_type(8)))  _Float16 v8h;
typedef __attribute__((ext_vector_type(8)))  float    v8f;
typedef __attribute__((ext_vector_type(4)))  float    v4f;

#define K_TAPS   27
#define CIN      16
#define COUT     4
#define NST      3
#define KTOT     432          // 27*16
#define KPAD     448          // 14*32
#define NKB      14           // K blocks of 32
#define BSTRIDE  456          // padded row stride (halfs) -> conflict-free ds reads
#define WAVES_PER_BLOCK 8

__global__ __launch_bounds__(256) void stencil_wmma_kernel(
    const float* __restrict__ x,
    const float* __restrict__ W,
    const long long* __restrict__ aprs,
    const long long* __restrict__ ldelta,
    float* __restrict__ out,
    int n, int ntiles)
{
    // Packed stencil matrix, column-major per output column:
    // Bcm[col][kk] = W[s][k][c][d] with col = 4*s + d, kk = k*16 + c.
    __shared__ __align__(16) _Float16 Bcm[16 * BSTRIDE];

    const int tid = threadIdx.x;
    for (int i = tid; i < 16 * KPAD; i += 256) {
        int col = i / KPAD;
        int kk  = i % KPAD;
        _Float16 v = (_Float16)0.0f;
        if (kk < KTOT && col < NST * COUT) {
            int s = col >> 2, d = col & 3;
            int k = kk >> 4, c = kk & 15;
            v = (_Float16)W[((s * K_TAPS + k) * CIN + c) * COUT + d];
        }
        Bcm[col * BSTRIDE + kk] = v;
    }
    __syncthreads();

    const int wave = tid >> 5;
    const int lane = tid & 31;
    const int rowm = lane & 15;   // A-matrix row (particle in tile) == C/D column index
    const int half = lane >> 4;   // half-wave selects K sub-block / M sub-block

    const int tile = blockIdx.x * WAVES_PER_BLOCK + wave;
    if (tile >= ntiles) return;
    const int tb = tile * 16;

    // This lane helps load particle (tb + rowm)'s A row.
    int pg = tb + rowm;
    if (pg >= n) pg = n - 1;      // tail clamp (N divides evenly; safety only)
    const long long* aprs_row = aprs + (long long)pg * K_TAPS;

    // Prefetch all 27 neighbor offsets as UNSIGNED 32-bit element offsets
    // (max N*16 < 2^31) so global loads use saddr + zext(voffset) form.
    // aprs is streamed exactly once -> non-temporal to keep x resident in L2.
    unsigned offs[K_TAPS];
#pragma unroll
    for (int k = 0; k < K_TAPS; ++k) {
        long long idx = __builtin_nontemporal_load(aprs_row + k);
        offs[k] = (unsigned)idx * (unsigned)CIN + (unsigned)(half * 8);
    }

    v8f acc = {};
    const _Float16* brow = &Bcm[rowm * BSTRIDE];   // column 'rowm' of packed B

#pragma unroll
    for (int kb = 0; kb < NKB; ++kb) {
        v16h a;
        // neighbor 2*kb -> K offsets {half*8 .. half*8+7} of this 32-block
        {
            const float* p = x + (unsigned long long)offs[2 * kb];
            v4f f0 = *(const v4f*)p;
            v4f f1 = *(const v4f*)(p + 4);
            a[0] = (_Float16)f0[0]; a[1] = (_Float16)f0[1];
            a[2] = (_Float16)f0[2]; a[3] = (_Float16)f0[3];
            a[4] = (_Float16)f1[0]; a[5] = (_Float16)f1[1];
            a[6] = (_Float16)f1[2]; a[7] = (_Float16)f1[3];
        }
        // neighbor 2*kb+1 -> K offsets {16+half*8 ..}; kb==13 is zero padding
        if (2 * kb + 1 < K_TAPS) {
            const float* p = x + (unsigned long long)offs[2 * kb + 1];
            v4f f0 = *(const v4f*)p;
            v4f f1 = *(const v4f*)(p + 4);
            a[8]  = (_Float16)f0[0]; a[9]  = (_Float16)f0[1];
            a[10] = (_Float16)f0[2]; a[11] = (_Float16)f0[3];
            a[12] = (_Float16)f1[0]; a[13] = (_Float16)f1[1];
            a[14] = (_Float16)f1[2]; a[15] = (_Float16)f1[3];
        } else {
            a[8]  = (_Float16)0.0f; a[9]  = (_Float16)0.0f;
            a[10] = (_Float16)0.0f; a[11] = (_Float16)0.0f;
            a[12] = (_Float16)0.0f; a[13] = (_Float16)0.0f;
            a[14] = (_Float16)0.0f; a[15] = (_Float16)0.0f;
        }

        // B fragment: 8 halfs at K = 32*kb + 8*half and 8 at K = 32*kb + 16 + 8*half (two 16B LDS reads)
        const _Float16* bp = brow + 32 * kb + 8 * half;
        v8h b0 = *(const v8h*)bp;
        v8h b1 = *(const v8h*)(bp + 16);
        v16h b = __builtin_shufflevector(b0, b1,
                 0, 1, 2, 3, 4, 5, 6, 7, 8, 9, 10, 11, 12, 13, 14, 15);

        acc = __builtin_amdgcn_wmma_f32_16x16x32_f16(
            /*neg_a=*/false, a, /*neg_b=*/false, b,
            /*c_mod=*/(short)0, acc, /*reuse_a=*/false, /*reuse_b=*/false);
    }

    // D layout: lane holds column N = rowm; VGPR r holds row M = r + 8*half.
    // Particle g selects stencil s: its outputs live in columns 4s..4s+3.
    // ldelta / out are single-pass streams -> non-temporal.
#pragma unroll
    for (int r = 0; r < 8; ++r) {
        int g = tb + r + 8 * half;
        if (g < n) {
            int s = (int)__builtin_nontemporal_load(ldelta + g);
            int d = rowm - 4 * s;
            if ((unsigned)d < 4u) {
                float v = acc[r];
                v = v > 0.0f ? v : 0.0f;
                __builtin_nontemporal_store(v, &out[(long long)g * COUT + d]);
            }
        }
    }
}

extern "C" void kernel_launch(void* const* d_in, const int* in_sizes, int n_in,
                              void* d_out, int out_size, void* d_ws, size_t ws_size,
                              hipStream_t stream) {
    const float*     x     = (const float*)d_in[0];
    const float*     W     = (const float*)d_in[1];
    const long long* aprs  = (const long long*)d_in[2];
    const long long* ld    = (const long long*)d_in[3];
    float*           out   = (float*)d_out;

    const int n      = in_sizes[0] / CIN;          // 1,000,000 particles
    const int ntiles = (n + 15) / 16;              // 62,500
    const int nblk   = (ntiles + WAVES_PER_BLOCK - 1) / WAVES_PER_BLOCK;

    stencil_wmma_kernel<<<dim3(nblk), dim3(256), 0, stream>>>(
        x, W, aprs, ld, out, n, ntiles);
}
